// ObjectClassifierMLP_33268816675157
// MI455X (gfx1250) — compile-verified, matrix operand order checked by reading, not run on hardware
//
#include <hip/hip_runtime.h>
#include <hip/hip_bf16.h>

// ---------------------------------------------------------------------------
// Types for WMMA
// ---------------------------------------------------------------------------
typedef __attribute__((ext_vector_type(16))) __bf16 v16bf;
typedef __attribute__((ext_vector_type(8)))  float  v8f;
typedef __attribute__((ext_vector_type(4)))  unsigned int u32x4;

union FragU { u32x4 q[2]; v16bf v; };

#define NROWS 65536
#define OBJD  2048
#define TAILW 352      // 200 (obj_embed) + 128 (pos) + 24 zero pad  -> K 2376->2400
#define KTOT  75       // 2400 / 32  (64 tiles from features, 11 from tail)
#define HID   1024
#define NCLS  37
#define LDSW  40       // bf16 LDS row stride: 80 bytes, 16B aligned, conflict-free

// float -> bf16 round-to-nearest-even
__device__ __forceinline__ unsigned int f2bf1(float f) {
    unsigned int u = __float_as_uint(f);
    return (u + 0x7FFFu + ((u >> 16) & 1u)) >> 16;
}
__device__ __forceinline__ unsigned int packbf2(float lo, float hi) {
    return f2bf1(lo) | (f2bf1(hi) << 16);
}

__device__ __forceinline__ void frag_load(const unsigned short* base, FragU& f) {
    f.q[0] = *(const u32x4*)(base);
    f.q[1] = *(const u32x4*)(base + 16);
}

// ---------------------------------------------------------------------------
// prep: transpose + convert dec_w1 [2376,1024] f32 -> w1t [1024][2400] bf16
// ---------------------------------------------------------------------------
__global__ void prep_w1_kernel(const float* __restrict__ w1, unsigned short* __restrict__ w1t) {
    __shared__ float tile[32][33];
    const int k0 = blockIdx.x * 32;   // 0..2368 (grid.x = 75)
    const int n0 = blockIdx.y * 32;   // 0..992  (grid.y = 32)
    const int tx = threadIdx.x, ty = threadIdx.y; // (32,8)
#pragma unroll
    for (int i = 0; i < 4; ++i) {
        int k = k0 + ty + i * 8;
        tile[ty + i * 8][tx] = (k < 2376) ? w1[(long)k * HID + (n0 + tx)] : 0.0f;
    }
    __syncthreads();
#pragma unroll
    for (int i = 0; i < 4; ++i) {
        int n = n0 + ty + i * 8;
        int k = k0 + tx;
        w1t[(long)n * 2400 + k] = (unsigned short)f2bf1(tile[tx][ty + i * 8]);
    }
}

// prep: dec_w2 [1024,37] f32 -> w2t [48][1024] bf16 (rows 37..47 zero)
__global__ void prep_w2_kernel(const float* __restrict__ w2, unsigned short* __restrict__ w2t) {
    int idx = blockIdx.x * 256 + threadIdx.x;   // 192 blocks -> 49152 = 48*1024
    int n = idx >> 10, k = idx & 1023;
    float v = (n < NCLS) ? w2[(long)k * NCLS + n] : 0.0f;
    w2t[idx] = (unsigned short)f2bf1(v);
}

// ---------------------------------------------------------------------------
// box stats: training BN over center_size(boxes[:,1:5]) -> scale4/shift4
// ---------------------------------------------------------------------------
__global__ void box_stats_kernel(const float* __restrict__ boxes,
                                 const float* __restrict__ g4, const float* __restrict__ b4,
                                 float* __restrict__ boxp /* [0..3]=scale, [4..7]=shift */) {
    __shared__ float red[256][8];
    const int tid = threadIdx.x;
    float s[4] = {0, 0, 0, 0}, q[4] = {0, 0, 0, 0};
    for (int row = tid; row < NROWS; row += 256) {
        const float* b = boxes + (long)row * 5;
        float x1 = b[1], y1 = b[2], x2 = b[3], y2 = b[4];
        float w = x2 - x1, h = y2 - y1;
        float cs[4] = {x1 + 0.5f * w, y1 + 0.5f * h, w, h};
#pragma unroll
        for (int j = 0; j < 4; ++j) { s[j] += cs[j]; q[j] += cs[j] * cs[j]; }
    }
#pragma unroll
    for (int j = 0; j < 4; ++j) { red[tid][j] = s[j]; red[tid][4 + j] = q[j]; }
    __syncthreads();
    for (int stride = 128; stride > 0; stride >>= 1) {
        if (tid < stride)
#pragma unroll
            for (int j = 0; j < 8; ++j) red[tid][j] += red[tid + stride][j];
        __syncthreads();
    }
    if (tid < 4) {
        float mean = red[0][tid] * (1.0f / NROWS);
        float var = red[0][4 + tid] * (1.0f / NROWS) - mean * mean;
        var = fmaxf(var, 0.0f);
        float sc = rsqrtf(var + 1e-5f) * g4[tid];
        boxp[tid] = sc;
        boxp[4 + tid] = b4[tid] - mean * sc;
    }
}

// ---------------------------------------------------------------------------
// tail builder: T[N,352] bf16 = [ dist@obj_w (200) | relu(bn(cs)@pos_w+pos_b) (128) | 0 (24) ]
// ---------------------------------------------------------------------------
__global__ __launch_bounds__(352) void build_tail_kernel(
    const float* __restrict__ dist, const float* __restrict__ boxes,
    const float* __restrict__ objw, const float* __restrict__ posw,
    const float* __restrict__ posb, const float* __restrict__ boxp,
    unsigned short* __restrict__ T) {
    __shared__ float s_objw[36 * 200];
    __shared__ float s_posw[4 * 128];
    __shared__ float s_posb[128];
    __shared__ float s_sc4[4], s_sh4[4];
    __shared__ float s_dist[36];
    __shared__ float s_cs[4];
    const int tid = threadIdx.x;
    for (int i = tid; i < 36 * 200; i += 352) s_objw[i] = objw[i];
    for (int i = tid; i < 512; i += 352) s_posw[i] = posw[i];
    if (tid < 128) s_posb[tid] = posb[tid];
    if (tid < 4) { s_sc4[tid] = boxp[tid]; s_sh4[tid] = boxp[4 + tid]; }
    __syncthreads();
    for (int rr = 0; rr < 16; ++rr) {
        long row = (long)blockIdx.x * 16 + rr;
        if (tid < 36) s_dist[tid] = dist[row * 36 + tid];
        if (tid < 4) {
            const float* b = boxes + row * 5;
            float cs;
            if (tid == 0)      cs = b[1] + 0.5f * (b[3] - b[1]);
            else if (tid == 1) cs = b[2] + 0.5f * (b[4] - b[2]);
            else if (tid == 2) cs = b[3] - b[1];
            else               cs = b[4] - b[2];
            s_cs[tid] = cs * s_sc4[tid] + s_sh4[tid];
        }
        __syncthreads();
        unsigned short val;
        if (tid < 200) {
            float a = 0.0f;
#pragma unroll
            for (int k = 0; k < 36; ++k) a += s_dist[k] * s_objw[k * 200 + tid];
            val = (unsigned short)f2bf1(a);
        } else if (tid < 328) {
            int j = tid - 200;
            float a = s_posb[j];
#pragma unroll
            for (int k = 0; k < 4; ++k) a += s_cs[k] * s_posw[k * 128 + j];
            val = (unsigned short)f2bf1(fmaxf(a, 0.0f));
        } else {
            val = 0;
        }
        T[row * TAILW + tid] = val;
        __syncthreads();
    }
}

// ---------------------------------------------------------------------------
// GEMM1: y[N,1024] = concat(features, tail) @ W1 + b1      (bf16 WMMA, f32 acc)
// grid (8, 512), block 256. Tile 128x128, K-tile 32 (= one wmma K).
// ---------------------------------------------------------------------------
__global__ __launch_bounds__(256) void gemm1_kernel(
    const float* __restrict__ features, const unsigned short* __restrict__ T,
    const unsigned short* __restrict__ w1t, const float* __restrict__ b1,
    float* __restrict__ y) {
    __shared__ __align__(16) unsigned short sA[2][128 * LDSW];
    __shared__ __align__(16) unsigned short sB[2][128 * LDSW];

    const int tid = threadIdx.x;
    const int bn = blockIdx.x, bm = blockIdx.y;
    const int r = tid >> 1;
    const int c0 = (tid & 1) << 4;  // 0 or 16
    const long arow = (long)(bm * 128 + r);

    const float* fptr = features + arow * OBJD + c0;
    const unsigned short* tptr = T + arow * TAILW + c0;
    const unsigned short* bptr = w1t + (long)(bn * 128 + r) * 2400 + c0;

    const int lane = tid & 31;
    const int wid = tid >> 5;
    const int wm = wid & 3;       // 4 waves along M (32 rows each)
    const int wn = wid >> 2;      // 2 waves along N (64 cols each)
    const int m16 = lane & 15;
    const int h = lane >> 4;

    v8f acc[2][4];
#pragma unroll
    for (int f = 0; f < 2; ++f)
#pragma unroll
        for (int g = 0; g < 4; ++g)
#pragma unroll
            for (int e = 0; e < 8; ++e) acc[f][g][e] = 0.0f;

    u32x4 a_rg[2], b_rg[2];

    // ---- tile loader (global -> regs) ----
    auto load_tile = [&](int kt) {
        if (kt < 64) {
            const float4* p = (const float4*)(fptr + kt * 32);
            float4 v0 = p[0], v1 = p[1], v2 = p[2], v3 = p[3];
            a_rg[0][0] = packbf2(v0.x, v0.y); a_rg[0][1] = packbf2(v0.z, v0.w);
            a_rg[0][2] = packbf2(v1.x, v1.y); a_rg[0][3] = packbf2(v1.z, v1.w);
            a_rg[1][0] = packbf2(v2.x, v2.y); a_rg[1][1] = packbf2(v2.z, v2.w);
            a_rg[1][2] = packbf2(v3.x, v3.y); a_rg[1][3] = packbf2(v3.z, v3.w);
            if (kt + 2 < 64) __builtin_prefetch(fptr + (kt + 2) * 32, 0, 1);
        } else {
            const u32x4* p = (const u32x4*)(tptr + (kt - 64) * 32);
            a_rg[0] = p[0]; a_rg[1] = p[1];
        }
        const u32x4* pb = (const u32x4*)(bptr + kt * 32);
        b_rg[0] = pb[0]; b_rg[1] = pb[1];
    };

    load_tile(0);
    int buf = 0;
    for (int kt = 0; kt < KTOT; ++kt) {
        // regs -> LDS
        u32x4* da = (u32x4*)&sA[buf][r * LDSW + c0];
        da[0] = a_rg[0]; da[1] = a_rg[1];
        u32x4* db = (u32x4*)&sB[buf][r * LDSW + c0];
        db[0] = b_rg[0]; db[1] = b_rg[1];
        __syncthreads();
        if (kt + 1 < KTOT) load_tile(kt + 1);

        const unsigned short* A = sA[buf];
        const unsigned short* B = sB[buf];
        FragU af[2], bfr[4];
#pragma unroll
        for (int f = 0; f < 2; ++f)
            frag_load(A + (wm * 32 + f * 16 + m16) * LDSW + h * 8, af[f]);
#pragma unroll
        for (int g = 0; g < 4; ++g)
            frag_load(B + (wn * 64 + g * 16 + m16) * LDSW + h * 8, bfr[g]);
#pragma unroll
        for (int f = 0; f < 2; ++f)
#pragma unroll
            for (int g = 0; g < 4; ++g)
                acc[f][g] = __builtin_amdgcn_wmma_f32_16x16x32_bf16(
                    false, af[f].v, false, bfr[g].v, (short)0, acc[f][g], false, false);
        buf ^= 1;
    }

    // epilogue: add bias, store y (fp32)
#pragma unroll
    for (int g = 0; g < 4; ++g) {
        int n = bn * 128 + wn * 64 + g * 16 + m16;
        float bias = b1[n];
#pragma unroll
        for (int f = 0; f < 2; ++f) {
#pragma unroll
            for (int r8 = 0; r8 < 8; ++r8) {
                int m = bm * 128 + wm * 32 + f * 16 + r8 + 8 * h;
                y[(long)m * HID + n] = acc[f][g][r8] + bias;
            }
        }
    }
}

// ---------------------------------------------------------------------------
// per-column partial sums/sumsq of y  (grid (4,256), block 256; no atomics)
// ---------------------------------------------------------------------------
__global__ void colstats_kernel(const float* __restrict__ y,
                                float* __restrict__ psum, float* __restrict__ psq) {
    int c = blockIdx.x * 256 + threadIdx.x;      // 0..1023
    long r0 = (long)blockIdx.y * 256;
    const float* p = y + r0 * HID + c;
    float s = 0.0f, q = 0.0f;
    for (int i = 0; i < 256; ++i) {
        float v = p[(long)i * HID];
        s += v; q += v * v;
    }
    psum[blockIdx.y * HID + c] = s;
    psq[blockIdx.y * HID + c] = q;
}

// reduce partials -> fused BN scale/shift per hidden column
__global__ void finstats_kernel(const float* __restrict__ psum, const float* __restrict__ psq,
                                const float* __restrict__ g, const float* __restrict__ b,
                                float* __restrict__ scale, float* __restrict__ shift) {
    int c = blockIdx.x * 256 + threadIdx.x;
    float s = 0.0f, q = 0.0f;
    for (int i = 0; i < 256; ++i) { s += psum[i * HID + c]; q += psq[i * HID + c]; }
    float mean = s * (1.0f / NROWS);
    float var = fmaxf(q * (1.0f / NROWS) - mean * mean, 0.0f);
    float sc = rsqrtf(var + 1e-5f) * g[c];
    scale[c] = sc;
    shift[c] = b[c] - mean * sc;
}

// ---------------------------------------------------------------------------
// GEMM2: out[N,37] = relu(y*scale+shift) @ W2 + b2   (bf16 WMMA, N padded 48)
// grid 512, block 256. Tile 128x48, K-tile 32.
// ---------------------------------------------------------------------------
__global__ __launch_bounds__(256) void gemm2_kernel(
    const float* __restrict__ y, const unsigned short* __restrict__ w2t,
    const float* __restrict__ scale, const float* __restrict__ shift,
    const float* __restrict__ b2, float* __restrict__ out) {
    __shared__ __align__(16) unsigned short sA[2][128 * LDSW];
    __shared__ __align__(16) unsigned short sB[2][48 * LDSW];

    const int tid = threadIdx.x;
    const int bm = blockIdx.x;
    const int r = tid >> 1;
    const int c0 = (tid & 1) << 4;
    const long arow = (long)(bm * 128 + r);
    const float* yptr = y + arow * HID + c0;

    const int lane = tid & 31;
    const int wid = tid >> 5;   // 8 waves, 16 rows each
    const int m16 = lane & 15;
    const int h = lane >> 4;

    v8f acc[3];
#pragma unroll
    for (int g = 0; g < 3; ++g)
#pragma unroll
        for (int e = 0; e < 8; ++e) acc[g][e] = 0.0f;

    u32x4 a_rg[2], b_rg[2];
    auto load_tile = [&](int kt) {
        const float4* p = (const float4*)(yptr + kt * 32);
        const float4* ps = (const float4*)(scale + kt * 32 + c0);
        const float4* ph = (const float4*)(shift + kt * 32 + c0);
#pragma unroll
        for (int i = 0; i < 4; ++i) {
            float4 v = p[i], sc = ps[i], sh = ph[i];
            float e0 = fmaxf(v.x * sc.x + sh.x, 0.0f);
            float e1 = fmaxf(v.y * sc.y + sh.y, 0.0f);
            float e2 = fmaxf(v.z * sc.z + sh.z, 0.0f);
            float e3 = fmaxf(v.w * sc.w + sh.w, 0.0f);
            a_rg[i >> 1][(i & 1) * 2 + 0] = packbf2(e0, e1);
            a_rg[i >> 1][(i & 1) * 2 + 1] = packbf2(e2, e3);
        }
        if (tid < 96) {
            const u32x4* pb = (const u32x4*)(w2t + (long)(tid >> 1) * HID + kt * 32 + ((tid & 1) << 4));
            b_rg[0] = pb[0]; b_rg[1] = pb[1];
        }
    };

    load_tile(0);
    int buf = 0;
    for (int kt = 0; kt < HID / 32; ++kt) {
        u32x4* da = (u32x4*)&sA[buf][r * LDSW + c0];
        da[0] = a_rg[0]; da[1] = a_rg[1];
        if (tid < 96) {
            u32x4* db = (u32x4*)&sB[buf][(tid >> 1) * LDSW + ((tid & 1) << 4)];
            db[0] = b_rg[0]; db[1] = b_rg[1];
        }
        __syncthreads();
        if (kt + 1 < HID / 32) load_tile(kt + 1);

        FragU af, bfr[3];
        frag_load(&sA[buf][(wid * 16 + m16) * LDSW + h * 8], af);
#pragma unroll
        for (int g = 0; g < 3; ++g)
            frag_load(&sB[buf][(g * 16 + m16) * LDSW + h * 8], bfr[g]);
#pragma unroll
        for (int g = 0; g < 3; ++g)
            acc[g] = __builtin_amdgcn_wmma_f32_16x16x32_bf16(
                false, af.v, false, bfr[g].v, (short)0, acc[g], false, false);
        buf ^= 1;
    }

#pragma unroll
    for (int g = 0; g < 3; ++g) {
        int n = g * 16 + m16;
        if (n < NCLS) {
            float bias = b2[n];
#pragma unroll
            for (int r8 = 0; r8 < 8; ++r8) {
                int m = bm * 128 + wid * 16 + r8 + 8 * h;
                out[(long)m * NCLS + n] = acc[g][r8] + bias;
            }
        }
    }
}

// ---------------------------------------------------------------------------
extern "C" void kernel_launch(void* const* d_in, const int* in_sizes, int n_in,
                              void* d_out, int out_size, void* d_ws, size_t ws_size,
                              hipStream_t stream) {
    (void)in_sizes; (void)n_in; (void)out_size; (void)ws_size;
    const float* dist     = (const float*)d_in[0];
    const float* boxes    = (const float*)d_in[1];
    const float* features = (const float*)d_in[2];
    const float* objw     = (const float*)d_in[3];
    const float* bn4_g    = (const float*)d_in[4];
    const float* bn4_b    = (const float*)d_in[5];
    const float* posw     = (const float*)d_in[6];
    const float* posb     = (const float*)d_in[7];
    const float* w1       = (const float*)d_in[8];
    const float* b1       = (const float*)d_in[9];
    const float* bn1_g    = (const float*)d_in[10];
    const float* bn1_b    = (const float*)d_in[11];
    const float* w2       = (const float*)d_in[12];
    const float* b2       = (const float*)d_in[13];
    float* out = (float*)d_out;
    char* ws = (char*)d_ws;

    // workspace layout (all offsets 256B-aligned)
    unsigned short* w1t  = (unsigned short*)(ws);                       // 1024*2400*2 = 4,915,200
    unsigned short* w2t  = (unsigned short*)(ws + 4915200);             // 48*1024*2   =    98,304
    float*          boxp = (float*)(ws + 5013504);                      // 256 B
    unsigned short* T    = (unsigned short*)(ws + 5013760);             // 65536*352*2 = 46,137,344
    float*          y    = (float*)(ws + 51151104);                     // 65536*1024*4 = 268,435,456
    float*          psum = (float*)(ws + 319586560);                    // 256*1024*4 = 1,048,576
    float*          psq  = (float*)(ws + 320635136);                    // 1,048,576
    float*          scl  = (float*)(ws + 321683712);                    // 4 KB
    float*          shf  = (float*)(ws + 321687808);                    // 4 KB

    prep_w1_kernel<<<dim3(75, 32), dim3(32, 8), 0, stream>>>(w1, w1t);
    prep_w2_kernel<<<192, 256, 0, stream>>>(w2, w2t);
    box_stats_kernel<<<1, 256, 0, stream>>>(boxes, bn4_g, bn4_b, boxp);
    build_tail_kernel<<<NROWS / 16, 352, 0, stream>>>(dist, boxes, objw, posw, posb, boxp, T);
    gemm1_kernel<<<dim3(8, 512), 256, 0, stream>>>(features, T, w1t, b1, y);
    colstats_kernel<<<dim3(4, 256), 256, 0, stream>>>(y, psum, psq);
    finstats_kernel<<<4, 256, 0, stream>>>(psum, psq, bn1_g, bn1_b, scl, shf);
    gemm2_kernel<<<512, 256, 0, stream>>>(y, w2t, scl, shf, b2, out);
}
